// DeepLSTM_38001870635449
// MI455X (gfx1250) — matrix operand order repaired and checked by reading, not compiled, not run on hardware
//
#include <hip/hip_runtime.h>
#include <hip/hip_bf16.h>

typedef __attribute__((ext_vector_type(16))) __bf16 v16bf;
typedef __attribute__((ext_vector_type(8)))  float  v8f;
typedef __attribute__((ext_vector_type(2)))  float  v2f;
typedef __attribute__((ext_vector_type(4)))  int    v4i;

#define SIZEU   512
#define DEPTH   3
#define BATCH   32
#define TSEQ    512
#define VOCAB   32000
#define KTOT    1536      // 3*SIZE
#define NGATES  2048      // 4*SIZE
#define NT_G    128       // 2048/16 n-tiles
#define KSTEPS  48        // 1536/32 k-steps
#define LDSROW  772       // dwords per staged row: 768 + 4 pad (bank-conflict pad)
#define NWG     16        // persistent workgroups
#define NLANES  (NWG*256)

// ---- workspace layout (bytes) ----
#define WS_SYNC   0            // 256 B  (arrive, gen)
#define WS_CST    256          // c state f32 [3][32][512] = 196608
#define WS_XBF    196864       // x bf16 [512][32][512]    = 16777216
#define WS_HST    16974080     // h bf16 [3][32][512]      = 98304
#define WS_GATES  17072384     // gates f32 [32][2048]     = 262144
#define WS_PICK   17334528     // picked f32 [32][1536]    = 196608
#define WS_WT     17531136     // W bf16 fragment-packed   = 18874368

__device__ __forceinline__ unsigned short f2bf_bits(float f) {
    unsigned u = __builtin_bit_cast(unsigned, f);
    return (unsigned short)((u + 0x7FFFu + ((u >> 16) & 1u)) >> 16);
}
__device__ __forceinline__ unsigned pk2(float a, float b) {
    return (unsigned)f2bf_bits(a) | ((unsigned)f2bf_bits(b) << 16);
}
__device__ __forceinline__ float sigf(float x)  { return 1.0f / (1.0f + __expf(-x)); }
__device__ __forceinline__ float tanhfst(float x) { return 2.0f * sigf(2.0f * x) - 1.0f; }

union AF { int4 q[2]; v16bf v; };

// ---- CDNA5 async global->LDS copy (ASYNCcnt path), with fallback ----
#if __has_builtin(__builtin_amdgcn_global_load_async_to_lds_b128)
#define HAVE_ASYNC_LDS 1
typedef __attribute__((address_space(1))) v4i v4i_as1;
typedef __attribute__((address_space(3))) v4i v4i_as3;
__device__ __forceinline__ void copy16_g2l(const void* g, void* lds) {
    v4i_as1* gp = (v4i_as1*)(unsigned long long)(uintptr_t)g;
    v4i_as3* lp = (v4i_as3*)(unsigned)(uintptr_t)lds;   // LDS offset = low 32 bits
    __builtin_amdgcn_global_load_async_to_lds_b128(gp, lp, 0, 0);
}
__device__ __forceinline__ void wait_async_lds() {
#if __has_builtin(__builtin_amdgcn_s_wait_asynccnt)
    __builtin_amdgcn_s_wait_asynccnt(0);
#else
    asm volatile("s_wait_asynccnt 0" ::: "memory");
#endif
}
#else
#define HAVE_ASYNC_LDS 0
__device__ __forceinline__ void copy16_g2l(const void* g, void* lds) {
    *(int4*)lds = *(const int4*)g;
}
__device__ __forceinline__ void wait_async_lds() {}
#endif

// ---------------- embedding gather + bf16 convert ----------------
__global__ void k_embed_gather(const int* __restrict__ tokens,
                               const float* __restrict__ emb,
                               unsigned short* __restrict__ xbf) {
    int ch = blockIdx.x * blockDim.x + threadIdx.x;           // 8 elems per thread
    int u8 = ch % (SIZEU / 8);
    int tb = ch / (SIZEU / 8);                                // t*32 + b
    int b  = tb % BATCH;
    int t  = tb / BATCH;
    int tok = tokens[b * TSEQ + t];
    const float* src = emb + (size_t)tok * SIZEU + u8 * 8;
    float4 lo = *(const float4*)src;
    float4 hi = *(const float4*)(src + 4);
    unsigned o0 = pk2(lo.x, lo.y), o1 = pk2(lo.z, lo.w);
    unsigned o2 = pk2(hi.x, hi.y), o3 = pk2(hi.z, hi.w);
    unsigned* dst = (unsigned*)(xbf + (size_t)tb * SIZEU + u8 * 8);
    dst[0] = o0; dst[1] = o1; dst[2] = o2; dst[3] = o3;
}

// ---------------- W_gates -> bf16 fragment-packed ----------------
// Slot s = ((l*128 + nt)*48 + ks)*32 + lane ; lane holds N = nt*16 + lane%16,
// K = ks*32 + (lane<16 ? 0 : 16) + e  for e = 0..15 (contiguous k order, same as the A staging).
__global__ void k_repack_w(const float* __restrict__ Wg,
                           unsigned short* __restrict__ Wt) {
    int s = blockIdx.x * blockDim.x + threadIdx.x;            // 589824 slots
    int lane = s & 31;
    int ks = (s >> 5) % KSTEPS;
    int nt = ((s >> 5) / KSTEPS) % NT_G;
    int l  = (s >> 5) / (KSTEPS * NT_G);
    int n  = nt * 16 + (lane & 15);
    int kb = ks * 32 + ((lane < 16) ? 0 : 16);
    const float* src = Wg + (size_t)l * KTOT * NGATES + n;
    unsigned* dst = (unsigned*)(Wt + (size_t)s * 16);
    #pragma unroll
    for (int e = 0; e < 16; e += 2) {
        float a = src[(size_t)(kb + e) * NGATES];
        float c = src[(size_t)(kb + e + 1) * NGATES];
        dst[e >> 1] = pk2(a, c);
    }
}

// ---------------- grid barrier over NWG workgroups ----------------
__device__ __forceinline__ void grid_barrier(unsigned* arrive, unsigned* gen,
                                             unsigned& localGen) {
    __threadfence();                 // release all prior global stores
    __syncthreads();
    if (threadIdx.x == 0) {
        unsigned g = localGen;
        unsigned prev = atomicAdd(arrive, 1u);
        if (prev == NWG - 1u) {
            atomicExch(arrive, 0u);
            atomicAdd(gen, 1u);
        } else {
            while (__hip_atomic_load(gen, __ATOMIC_ACQUIRE,
                                     __HIP_MEMORY_SCOPE_AGENT) <= g) {
                __builtin_amdgcn_s_sleep(2);
            }
        }
    }
    localGen++;
    __syncthreads();
    __threadfence();                 // acquire: invalidate stale WGP$ lines
}

// ---------------- persistent LSTM scan ----------------
__global__ void k_lstm_scan(const float* __restrict__ bg,
                            const int* __restrict__ nstarts,
                            unsigned char* __restrict__ ws) {
    __shared__ unsigned cell[32 * LDSROW];                    // cell_in bf16 [32][1536]+pad

    unsigned* arrive = (unsigned*)(ws + WS_SYNC);
    unsigned* gen    = arrive + 1;
    float*          cstate = (float*)(ws + WS_CST);
    unsigned short* xbf    = (unsigned short*)(ws + WS_XBF);
    unsigned short* hstate = (unsigned short*)(ws + WS_HST);
    float*          gates  = (float*)(ws + WS_GATES);
    float*          picked = (float*)(ws + WS_PICK);
    const unsigned short* Wt = (const unsigned short*)(ws + WS_WT);

    const int tid  = threadIdx.x;
    const int gtid = blockIdx.x * 256 + tid;                  // 0..4095
    const int wid  = gtid >> 5;                               // 0..127 == n-tile
    const int lane = tid & 31;
    const int kadd = (lane < 16) ? 0 : 8;                     // A-frag dword offset
    const int mrow = lane & 15;
    unsigned localGen = 0;

    for (int t = 0; t < TSEQ; ++t) {
        const unsigned short* xt = xbf + (size_t)t * BATCH * SIZEU;
        for (int l = 0; l < DEPTH; ++l) {
            // ---- stage cell_in = [skip | cur | h_l] (bf16) into LDS ----
            for (int ch = tid; ch < 32 * (KTOT / 8); ch += 256) {
                int m  = ch / (KTOT / 8);
                int K0 = (ch % (KTOT / 8)) * 8;
                unsigned* ldst = &cell[m * LDSROW + (K0 >> 1)];
                if (K0 < SIZEU) {                             // skip region
                    if (l == 0) { *(int4*)ldst = make_int4(0, 0, 0, 0); continue; }
                    copy16_g2l(xt + m * SIZEU + K0, ldst);
                } else if (K0 < 2 * SIZEU) {                  // cur region
                    int off = K0 - SIZEU;
                    const void* s = (l == 0)
                        ? (const void*)(xt + m * SIZEU + off)
                        : (const void*)(hstate + ((l - 1) * BATCH + m) * SIZEU + off);
                    copy16_g2l(s, ldst);
                } else {                                      // own h region
                    int off = K0 - 2 * SIZEU;
                    copy16_g2l(hstate + (l * BATCH + m) * SIZEU + off, ldst);
                }
            }
            wait_async_lds();
            __syncthreads();

            // ---- gate GEMM: wave owns n-tile `wid`, both M-tiles ----
            v8f acc0 = {}, acc1 = {};
            const unsigned short* wt =
                Wt + ((size_t)(l * NT_G + wid) * KSTEPS) * 512;
            for (int ks = 0; ks < KSTEPS; ++ks) {
                AF a0, a1, bb;
                int b0 = mrow * LDSROW + ks * 16 + kadd;
                a0.q[0] = *(const int4*)&cell[b0];
                a0.q[1] = *(const int4*)&cell[b0 + 4];
                int b1 = (mrow + 16) * LDSROW + ks * 16 + kadd;
                a1.q[0] = *(const int4*)&cell[b1];
                a1.q[1] = *(const int4*)&cell[b1 + 4];
                const unsigned short* wp = wt + (size_t)ks * 512 + lane * 16;
                bb.q[0] = *(const int4*)wp;
                bb.q[1] = *(const int4*)(wp + 8);
                acc0 = __builtin_amdgcn_wmma_f32_16x16x32_bf16(
                    false, a0.v, false, bb.v, (short)0, acc0, false, false);
                acc1 = __builtin_amdgcn_wmma_f32_16x16x32_bf16(
                    false, a1.v, false, bb.v, (short)0, acc1, false, false);
            }
            // epilogue: + bias, store gates f32 [32][2048]
            int n = wid * 16 + (lane & 15);
            float bias = bg[l * NGATES + n];
            int mb = (lane < 16) ? 0 : 8;
            #pragma unroll
            for (int r = 0; r < 8; ++r) {
                gates[(mb + r) * NGATES + n]        = acc0[r] + bias;
                gates[(16 + mb + r) * NGATES + n]   = acc1[r] + bias;
            }
            grid_barrier(arrive, gen, localGen);

            // ---- pointwise LSTM cell (i, j, f, o) ----
            for (int idx = gtid; idx < BATCH * SIZEU; idx += NLANES) {
                int b = idx >> 9;
                int u = idx & (SIZEU - 1);
                float gi = gates[b * NGATES + u];
                float gj = gates[b * NGATES + SIZEU + u];
                float gf = gates[b * NGATES + 2 * SIZEU + u];
                float go = gates[b * NGATES + 3 * SIZEU + u];
                float* cp = cstate + (l * BATCH + b) * SIZEU + u;
                float c  = *cp;
                float cn = sigf(gf) * c + sigf(gi) * tanhfst(gj);
                float hn = sigf(go) * tanhfst(cn);
                *cp = cn;
                hstate[(l * BATCH + b) * SIZEU + u] = f2bf_bits(hn);
                if (t == nstarts[b])
                    picked[b * (DEPTH * SIZEU) + l * SIZEU + u] = hn;
            }
            grid_barrier(arrive, gen, localGen);
        }
    }
}

// ---------------- output projection: picked @ W_out^T ----------------
__global__ void k_proj(const float* __restrict__ Wout,
                       const unsigned char* __restrict__ ws,
                       float* __restrict__ out) {
    __shared__ unsigned pk[32 * LDSROW];
    const float* picked = (const float*)(ws + WS_PICK);
    const int tid = threadIdx.x;
    // stage picked (f32 -> bf16) as A-source rows in LDS
    for (int e2 = tid; e2 < BATCH * (KTOT / 2); e2 += 256) {
        int m  = e2 / (KTOT / 2);
        int k2 = e2 % (KTOT / 2);
        const float* s = picked + m * KTOT + k2 * 2;
        pk[m * LDSROW + k2] = pk2(s[0], s[1]);
    }
    __syncthreads();

    const int wave = tid >> 5, lane = tid & 31;
    const int kadd = (lane < 16) ? 0 : 8;
    const int mrow = lane & 15;
    const int vrow = (lane & 15);                 // N column within tile
    const int kb0  = (lane < 16) ? 0 : 16;

    for (int nt = blockIdx.x * 8 + wave; nt < VOCAB / 16; nt += gridDim.x * 8) {
        v8f acc0 = {}, acc1 = {};
        const float* wr0 = Wout + (size_t)(nt * 16 + vrow) * KTOT + kb0;
        for (int ks = 0; ks < KSTEPS; ++ks) {
            AF a0, a1, bb;
            int b0 = mrow * LDSROW + ks * 16 + kadd;
            a0.q[0] = *(const int4*)&pk[b0];
            a0.q[1] = *(const int4*)&pk[b0 + 4];
            int b1 = (mrow + 16) * LDSROW + ks * 16 + kadd;
            a1.q[0] = *(const int4*)&pk[b1];
            a1.q[1] = *(const int4*)&pk[b1 + 4];
            const float* wr = wr0 + ks * 32;
            unsigned* bu = (unsigned*)&bb;
            #pragma unroll
            for (int j = 0; j < 8; ++j) {
                v2f f = __builtin_nontemporal_load((const v2f*)(wr + 2 * j));
                bu[j] = pk2(f.x, f.y);
            }
            acc0 = __builtin_amdgcn_wmma_f32_16x16x32_bf16(
                false, a0.v, false, bb.v, (short)0, acc0, false, false);
            acc1 = __builtin_amdgcn_wmma_f32_16x16x32_bf16(
                false, a1.v, false, bb.v, (short)0, acc1, false, false);
        }
        int n  = nt * 16 + (lane & 15);
        int mb = (lane < 16) ? 0 : 8;
        #pragma unroll
        for (int r = 0; r < 8; ++r) {
            out[(mb + r) * VOCAB + n]      = acc0[r];
            out[(16 + mb + r) * VOCAB + n] = acc1[r];
        }
    }
}

extern "C" void kernel_launch(void* const* d_in, const int* in_sizes, int n_in,
                              void* d_out, int out_size, void* d_ws, size_t ws_size,
                              hipStream_t stream) {
    (void)in_sizes; (void)n_in; (void)out_size; (void)ws_size;
    const int*   tokens  = (const int*)d_in[0];
    const int*   nstarts = (const int*)d_in[1];
    const float* emb     = (const float*)d_in[2];
    const float* Wg      = (const float*)d_in[3];
    const float* bg      = (const float*)d_in[4];
    const float* Wout    = (const float*)d_in[5];
    unsigned char* ws = (unsigned char*)d_ws;

    // zero: sync block + c-state (contiguous), and h-state (bf16)
    (void)hipMemsetAsync(ws + WS_SYNC, 0, WS_XBF, stream);
    (void)hipMemsetAsync(ws + WS_HST, 0, DEPTH * BATCH * SIZEU * 2, stream);

    k_embed_gather<<<dim3(4096), dim3(256), 0, stream>>>(
        tokens, emb, (unsigned short*)(ws + WS_XBF));
    k_repack_w<<<dim3(2304), dim3(256), 0, stream>>>(
        Wg, (unsigned short*)(ws + WS_WT));
    k_lstm_scan<<<dim3(NWG), dim3(256), 0, stream>>>(bg, nstarts, ws);
    k_proj<<<dim3(250), dim3(256), 0, stream>>>(Wout, ws, (float*)d_out);
}
